// BidirLinearAttentionLayer_25409026523897
// MI455X (gfx1250) — matrix operand order repaired and checked by reading, not compiled, not run on hardware
//
#include <hip/hip_runtime.h>
#include <hip/hip_bf16.h>

// ---------------------------------------------------------------------------
// BidirLinearAttentionLayer for MI455X (gfx1250, wave32, WMMA)
// B=2, T=2048, D=512, H=8, HD=64, FF=2048
// All GEMMs consume pre-transposed f16 weights (N x K), so every WMMA operand
// is a pair of contiguous 128-bit loads (contiguous-k fragment layouts).
// ---------------------------------------------------------------------------

typedef __attribute__((ext_vector_type(16))) _Float16 v16h;
typedef __attribute__((ext_vector_type(8)))  float    v8f;

#define WMMA_F16(a, b, c) \
  __builtin_amdgcn_wmma_f32_16x16x32_f16(false, (a), false, (b), (short)0, (c), false, false)

static constexpr int  kB  = 2;
static constexpr int  kT  = 2048;
static constexpr int  kD  = 512;
static constexpr int  kH  = 8;
static constexpr int  kHD = 64;
static constexpr int  kFF = 2048;
static constexpr int  kBT = kB * kT;   // 4096 rows

// ---------------------------------------------------------------------------
// Transpose + f32->f16 convert for weights: in R x C (f32) -> out C x R (f16)
// ---------------------------------------------------------------------------
__global__ void transpose_f32_to_f16_kernel(const float* __restrict__ in,
                                            _Float16* __restrict__ out,
                                            int R, int C) {
  __shared__ float tile[32][33];
  const int c0 = blockIdx.x * 32, r0 = blockIdx.y * 32;
  const int tx = threadIdx.x, ty = threadIdx.y;   // 32 x 8
#pragma unroll
  for (int i = 0; i < 32; i += 8) {
    tile[ty + i][tx] = in[(size_t)(r0 + ty + i) * C + c0 + tx];
  }
  __syncthreads();
#pragma unroll
  for (int i = 0; i < 32; i += 8) {
    out[(size_t)(c0 + ty + i) * R + r0 + tx] = (_Float16)tile[tx][ty + i];
  }
}

// ---------------------------------------------------------------------------
// LayerNorm over D=512, one block (256 threads) per row, f16 output
// ---------------------------------------------------------------------------
__global__ void layernorm_f16_kernel(const float* __restrict__ x,
                                     const float* __restrict__ g,
                                     const float* __restrict__ bb,
                                     _Float16* __restrict__ out) {
  __shared__ float sred[256];
  const int tid = threadIdx.x;
  const int row = blockIdx.x;
  const float* xr = x + (size_t)row * kD;

  float v0 = xr[tid];
  float v1 = xr[tid + 256];

  sred[tid] = v0 + v1;
  __syncthreads();
  for (int st = 128; st > 0; st >>= 1) {
    if (tid < st) sred[tid] += sred[tid + st];
    __syncthreads();
  }
  const float mu = sred[0] * (1.0f / kD);
  __syncthreads();

  const float d0 = v0 - mu, d1 = v1 - mu;
  sred[tid] = d0 * d0 + d1 * d1;
  __syncthreads();
  for (int st = 128; st > 0; st >>= 1) {
    if (tid < st) sred[tid] += sred[tid + st];
    __syncthreads();
  }
  const float rs = rsqrtf(sred[0] * (1.0f / kD) + 1e-5f);

  _Float16* orow = out + (size_t)row * kD;
  orow[tid]       = (_Float16)(d0 * rs * g[tid]       + bb[tid]);
  orow[tid + 256] = (_Float16)(d1 * rs * g[tid + 256] + bb[tid + 256]);
}

// ---------------------------------------------------------------------------
// Fragment loaders (contiguous-k 16-bit WMMA layouts, wave32)
// A 16x32: lanes 0-15 hold M=lane, K 0..15; lanes 16-31 K 16..31
// ---------------------------------------------------------------------------
__device__ inline v16h load_a_frag(const _Float16* __restrict__ A, int lda, int l) {
  union { v16h v; uint4 u[2]; } f;
  const int m = l & 15;
  const int off = (l < 16) ? 0 : 16;
  const _Float16* row = A + (size_t)m * lda;
  f.u[0] = *(const uint4*)(row + off);
  f.u[1] = *(const uint4*)(row + off + 8);
  return f.v;
}

// B 32x16 fragment where the K dimension is CONTIGUOUS in memory with leading
// dim `ldn` between columns: B[k][n] = src[n*ldn + k]. Lanes 0-15: K=0..15,
// lanes 16-31: K=16..31 -> two contiguous 16-byte chunks per lane.
__device__ inline v16h load_bT_frag(const _Float16* __restrict__ src, int ldn, int l) {
  union { v16h v; uint4 u[2]; } f;
  const int n = l & 15;
  const int kb = (l < 16) ? 0 : 16;
  const _Float16* p = src + (size_t)n * ldn + kb;
  f.u[0] = *(const uint4*)(p);
  f.u[1] = *(const uint4*)(p + 8);
  return f.v;
}

// ---------------------------------------------------------------------------
// WMMA GEMM, no LDS: C[MxN] = A[MxK](f16) * Bt^T (+bias)(+epilogue)
//   A : M x K row-major f16
//   Bt: N x K row-major f16 (pre-transposed weights) -> frags are b128 pairs
// Block = 128 threads (4 waves), block tile 128(M) x 64(N).
// Each wave computes 2 M-tiles x 4 N-tiles (32x64): 8 WMMAs per K-step,
// each B fragment reused 2x, each A fragment 4x.
// EPI: 0 = none, 1 = ELU+1 feature map, 2 = exact GELU
// REORDER: 0 = row-major (ld=N), 1 = (B,H,T,HD), 2 = (B,H,HD,T)
// ---------------------------------------------------------------------------
template <int EPI, int REORDER>
__global__ void gemm_wmma_kernel(const _Float16* __restrict__ A,
                                 const _Float16* __restrict__ Bt,
                                 const float* __restrict__ bias,
                                 const float* __restrict__ resid,
                                 const unsigned char* __restrict__ maskBT,
                                 float* __restrict__ outF,
                                 _Float16* __restrict__ outH,
                                 int M, int N, int K) {
  const int l     = threadIdx.x & 31;
  const int wave  = threadIdx.x >> 5;
  const int nBase = blockIdx.x * 64;
  const int mBase = blockIdx.y * 128 + wave * 32;   // two 16-row tiles

  const _Float16* A0 = A + (size_t)mBase * K;
  const _Float16* A1 = A0 + (size_t)16 * K;
  const _Float16* Bb = Bt + (size_t)nBase * K;

  v8f acc[2][4] = {};

  for (int kk = 0; kk < K; kk += 32) {
    const v16h fa0 = load_a_frag(A0 + kk, K, l);
    const v16h fa1 = load_a_frag(A1 + kk, K, l);
    if (kk + 32 < K) {
      __builtin_prefetch(A0 + (size_t)(l & 15) * K + kk + 32, 0, 3);  // global_prefetch_b8
      __builtin_prefetch(Bb + (size_t)(l & 15) * K + kk + 32, 0, 3);
    }
#pragma unroll
    for (int nt = 0; nt < 4; ++nt) {
      const v16h fb = load_bT_frag(Bb + (size_t)(nt * 16) * K + kk, K, l);
      acc[0][nt] = WMMA_F16(fa0, fb, acc[0][nt]);
      acc[1][nt] = WMMA_F16(fa1, fb, acc[1][nt]);
    }
  }

  // ---- epilogue ----
  const int rOff = (l >= 16) ? 8 : 0;
#pragma unroll
  for (int mt = 0; mt < 2; ++mt) {
#pragma unroll
    for (int nt = 0; nt < 4; ++nt) {
      const int col = nBase + nt * 16 + (l & 15);
      const float bcol = bias ? bias[col] : 0.0f;
#pragma unroll
      for (int r = 0; r < 8; ++r) {
        const int row = mBase + mt * 16 + r + rOff;
        float v = acc[mt][nt][r] + bcol;
        if (EPI == 1) {
          v = (v > 0.0f) ? (v + 1.0f) : __expf(v);               // elu(v)+1
        } else if (EPI == 2) {
          v = 0.5f * v * (1.0f + erff(v * 0.70710678118654752f));
        }
        if (maskBT && maskBT[row]) v = 0.0f;
        if (resid) v += resid[(size_t)row * N + col];
        if (outF) outF[(size_t)row * N + col] = v;
        if (outH) {
          size_t idx;
          if (REORDER == 1) {          // (B,H,T,HD)
            const int b_ = row >> 11, t_ = row & (kT - 1);
            const int h_ = col >> 6,  d_ = col & (kHD - 1);
            idx = ((((size_t)b_ * kH) + h_) * kT + t_) * kHD + d_;
          } else if (REORDER == 2) {   // (B,H,HD,T)  transposed V
            const int b_ = row >> 11, t_ = row & (kT - 1);
            const int h_ = col >> 6,  d_ = col & (kHD - 1);
            idx = ((((size_t)b_ * kH) + h_) * kHD + d_) * kT + t_;
          } else {
            idx = (size_t)row * N + col;
          }
          outH[idx] = (_Float16)v;
        }
      }
    }
  }
}

// ---------------------------------------------------------------------------
// Decay-masked attention. One wave per (b,h, 16-row Q tile), j stepped by 32.
// (Single Q tile per wave: keeps live fragments ~130 VGPRs, no spills; the
//  2-tile variant spilled Q frags to scratch inside the hot loop.)
// Q,K: (B,H,T,HD) f16;  Vt: (B,H,HD,T) f16 (transposed). Out: (B,T,H*HD) f16.
//   S = (Q K^T) * decay^|i-j|;  out = (S / rowsum(S)) V
// ---------------------------------------------------------------------------
__global__ void __launch_bounds__(32, 1)
attn_decay_kernel(const _Float16* __restrict__ Q,
                  const _Float16* __restrict__ Kh,
                  const _Float16* __restrict__ Vt,
                  const float* __restrict__ decay_logit,
                  _Float16* __restrict__ attnOut) {
  __shared__ __align__(16) _Float16 sS[16 * 32];   // S~ tile, row-major 16x32

  const int l      = threadIdx.x;              // 0..31
  const int tilesT = kT / 16;                  // 128
  const int bh     = blockIdx.x / tilesT;      // 0..15
  const int itile  = blockIdx.x % tilesT;
  const int b      = bh / kH;
  const int h      = bh % kH;

  // decay = sigmoid(logit); logd = log(clip(decay, 1e-8))
  const float dl   = decay_logit[h];
  const float dec  = 1.0f / (1.0f + __expf(-dl));
  const float logd = __logf(fmaxf(dec, 1e-8f));

  const _Float16* Qb = Q + ((size_t)bh * kT + itile * 16) * kHD;
  const v16h aq0 = load_a_frag(Qb,      kHD, l);
  const v16h aq1 = load_a_frag(Qb + 32, kHD, l);

  const _Float16* Vbh = Vt + (size_t)bh * kHD * kT;   // [d][t], ld = kT

  v8f   oacc[4] = {};
  float rpart[8] = {};

  const int rOff = (l >= 16) ? 8 : 0;
  const int nLoc = l & 15;
  const int mGlb = itile * 16 + rOff;          // + r inside the loop

  for (int j0 = 0; j0 < kT; j0 += 32) {
    const _Float16* Kb0 = Kh + ((size_t)bh * kT + j0) * kHD;
    const _Float16* Kb1 = Kb0 + 16 * kHD;

    // Two 16x16 S tiles = Q(16x64) * K^T(64x16)   (K^T frags are contiguous)
    v8f s0 = {}; v8f s1 = {};
    s0 = WMMA_F16(aq0, load_bT_frag(Kb0,      kHD, l), s0);
    s0 = WMMA_F16(aq1, load_bT_frag(Kb0 + 32, kHD, l), s0);
    s1 = WMMA_F16(aq0, load_bT_frag(Kb1,      kHD, l), s1);
    s1 = WMMA_F16(aq1, load_bT_frag(Kb1 + 32, kHD, l), s1);

    // decay weight + rowsum partials; park S~(16x32) in LDS (C->A re-layout)
    __syncthreads();
#pragma unroll
    for (int r = 0; r < 8; ++r) {
      const int m  = mGlb + r;
      const float w0 = __expf(fabsf((float)(m - (j0 + nLoc)))      * logd);
      const float w1 = __expf(fabsf((float)(m - (j0 + 16 + nLoc))) * logd);
      const float sv0 = s0[r] * w0;
      const float sv1 = s1[r] * w1;
      rpart[r] += sv0 + sv1;
      sS[(r + rOff) * 32 + nLoc]      = (_Float16)sv0;
      sS[(r + rOff) * 32 + 16 + nLoc] = (_Float16)sv1;
    }
    __syncthreads();

    // A fragment of S~ (full 16x32, contiguous-k A layout) from LDS
    union { v16h v; uint4 u[2]; } fs;
    {
      const _Float16* p = sS + (size_t)(l & 15) * 32 + ((l < 16) ? 0 : 16);
      fs.u[0] = *(const uint4*)(p);
      fs.u[1] = *(const uint4*)(p + 8);
    }

    // O += S~(16x32) * V(32x64); V^T layout makes frags contiguous b128 pairs
#pragma unroll
    for (int c = 0; c < 4; ++c) {
      union { v16h v; uint4 u[2]; } fv;
      const _Float16* p = Vbh + (size_t)(c * 16 + nLoc) * kT + j0 + ((l < 16) ? 0 : 16);
      fv.u[0] = *(const uint4*)(p);
      fv.u[1] = *(const uint4*)(p + 8);
      oacc[c] = WMMA_F16(fs.v, fv.v, oacc[c]);
    }
  }

  // normalize by rowsum (reduce across the 16 lanes holding each row) + store
#pragma unroll
  for (int r = 0; r < 8; ++r) {
    float sum = rpart[r];
    sum += __shfl_xor(sum, 1);
    sum += __shfl_xor(sum, 2);
    sum += __shfl_xor(sum, 4);
    sum += __shfl_xor(sum, 8);
    const float inv = 1.0f / fmaxf(sum, 1e-6f);
    const int m = mGlb + r;
    _Float16* orow = attnOut + ((size_t)b * kT + m) * kD + h * kHD;
#pragma unroll
    for (int c = 0; c < 4; ++c) {
      orow[c * 16 + nLoc] = (_Float16)(oacc[c][r] * inv);
    }
  }
}

// ---------------------------------------------------------------------------
// Launch
// ---------------------------------------------------------------------------
extern "C" void kernel_launch(void* const* d_in, const int* in_sizes, int n_in,
                              void* d_out, int out_size, void* d_ws, size_t ws_size,
                              hipStream_t stream) {
  const float*         x     = (const float*)d_in[0];
  const unsigned char* mask  = (const unsigned char*)d_in[1];
  const float*         wq    = (const float*)d_in[2];
  const float*         wk    = (const float*)d_in[3];
  const float*         wv    = (const float*)d_in[4];
  const float*         wo    = (const float*)d_in[5];
  const float*         bo    = (const float*)d_in[6];
  const float*         g1    = (const float*)d_in[7];
  const float*         b1    = (const float*)d_in[8];
  const float*         g2    = (const float*)d_in[9];
  const float*         b2    = (const float*)d_in[10];
  const float*         w1    = (const float*)d_in[11];
  const float*         bf1   = (const float*)d_in[12];
  const float*         w2    = (const float*)d_in[13];
  const float*         bf2   = (const float*)d_in[14];
  const float*         dcl   = (const float*)d_in[15];

  char* ws = (char*)d_ws;
  size_t off = 0;
  auto carve = [&](size_t bytes) -> void* {
    void* p = ws + off;
    off = (off + bytes + 255) & ~(size_t)255;
    return p;
  };

  _Float16* xn1_h  = (_Float16*)carve((size_t)kBT * kD * 2);
  _Float16* wqT    = (_Float16*)carve((size_t)kD * kD * 2);    // N x K
  _Float16* wkT    = (_Float16*)carve((size_t)kD * kD * 2);
  _Float16* wvT    = (_Float16*)carve((size_t)kD * kD * 2);
  _Float16* woT    = (_Float16*)carve((size_t)kD * kD * 2);
  _Float16* w1T    = (_Float16*)carve((size_t)kFF * kD * 2);   // FF x D
  _Float16* w2T    = (_Float16*)carve((size_t)kD * kFF * 2);   // D x FF
  _Float16* Qh     = (_Float16*)carve((size_t)kB * kH * kT * kHD * 2);
  _Float16* Kh2    = (_Float16*)carve((size_t)kB * kH * kT * kHD * 2);
  _Float16* Vth    = (_Float16*)carve((size_t)kB * kH * kHD * kT * 2);
  _Float16* attn_h = (_Float16*)carve((size_t)kBT * kD * 2);
  float*    x2_f   = (float*)   carve((size_t)kBT * kD * 4);
  _Float16* xn2_h  = (_Float16*)carve((size_t)kBT * kD * 2);
  _Float16* ff_h   = (_Float16*)carve((size_t)kBT * kFF * 2);
  (void)ws_size; (void)n_in; (void)in_sizes; (void)out_size;

  // 1) Weight transpose+convert to f16 (once per launch; tiny vs GEMMs)
  {
    dim3 blk(32, 8);
    auto tcvt = [&](const float* src, _Float16* dst, int R, int C) {
      dim3 grd(C / 32, R / 32);
      transpose_f32_to_f16_kernel<<<grd, blk, 0, stream>>>(src, dst, R, C);
    };
    tcvt(wq, wqT, kD, kD);
    tcvt(wk, wkT, kD, kD);
    tcvt(wv, wvT, kD, kD);
    tcvt(wo, woT, kD, kD);
    tcvt(w1, w1T, kD, kFF);
    tcvt(w2, w2T, kFF, kD);
  }

  // 2) LN1
  layernorm_f16_kernel<<<kBT, 256, 0, stream>>>(x, g1, b1, xn1_h);

  // 3) Q/K/V projections (f16 WMMA, no-LDS, transposed weights)
  {
    dim3 grid(kD / 64, kBT / 128), block(128);
    // Q = elu(xn@wq)+1                           -> (B,H,T,HD)
    gemm_wmma_kernel<1, 1><<<grid, block, 0, stream>>>(xn1_h, wqT, nullptr,
        nullptr, nullptr, nullptr, Qh, kBT, kD, kD);
    // K = (elu(xn@wk)+1) * pad                   -> (B,H,T,HD)
    gemm_wmma_kernel<1, 1><<<grid, block, 0, stream>>>(xn1_h, wkT, nullptr,
        nullptr, mask, nullptr, Kh2, kBT, kD, kD);
    // V = (xn@wv) * pad                          -> (B,H,HD,T) transposed
    gemm_wmma_kernel<0, 2><<<grid, block, 0, stream>>>(xn1_h, wvT, nullptr,
        nullptr, mask, nullptr, Vth, kBT, kD, kD);
  }

  // 4) Decay-masked attention (16 Q rows per wave, spill-free)
  attn_decay_kernel<<<kB * kH * (kT / 16), 32, 0, stream>>>(Qh, Kh2, Vth, dcl, attn_h);

  // 5) Output projection + residual: x2 = x + attn@wo + bo
  {
    dim3 grid(kD / 64, kBT / 128), block(128);
    gemm_wmma_kernel<0, 0><<<grid, block, 0, stream>>>(attn_h, woT, bo, x,
        nullptr, x2_f, nullptr, kBT, kD, kD);
  }

  // 6) LN2
  layernorm_f16_kernel<<<kBT, 256, 0, stream>>>(x2_f, g2, b2, xn2_h);

  // 7) FFN1: gelu(xn2@w1 + bf1)
  {
    dim3 grid(kFF / 64, kBT / 128), block(128);
    gemm_wmma_kernel<2, 0><<<grid, block, 0, stream>>>(xn2_h, w1T, bf1,
        nullptr, nullptr, nullptr, ff_h, kBT, kFF, kD);
  }

  // 8) FFN2 + residual -> final output: out = x2 + ff@w2 + bf2
  {
    dim3 grid(kD / 64, kBT / 128), block(128);
    gemm_wmma_kernel<0, 0><<<grid, block, 0, stream>>>(ff_h, w2T, bf2, x2_f,
        nullptr, (float*)d_out, nullptr, kBT, kD, kFF);
  }
}
